// EdgePartnerNetwork_87608742904296
// MI455X (gfx1250) — matrix operand order repaired and checked by reading, not compiled, not run on hardware
//
#include <hip/hip_runtime.h>

typedef __attribute__((ext_vector_type(16))) __bf16       v16bf;
typedef __attribute__((ext_vector_type(8)))  float        v8f;
typedef __attribute__((ext_vector_type(8)))  unsigned int v8u;

#define DIM    128   // node / graph embedding dim (K per GEMM piece)
#define HDIM   256   // hidden dim (N of GEMM)
#define NTILE  16    // 16 column tiles of 16 -> 256 columns
#define HTILE  8     // tiles processed per half (acc live range = 64 VGPRs)
#define KSTEPS 4     // 128 K / 32 per WMMA

// f32 pair -> packed bf16 dword (low = first element), round-to-nearest-even
__device__ __forceinline__ unsigned pk_bf16(float a, float b) {
#if __has_builtin(__builtin_amdgcn_cvt_pk_bf16_f32)
    return __builtin_bit_cast(unsigned, __builtin_amdgcn_cvt_pk_bf16_f32(a, b));
#else
    unsigned ua = __float_as_uint(a);
    unsigned ub = __float_as_uint(b);
    ua += 0x7FFFu + ((ua >> 16) & 1u);
    ub += 0x7FFFu + ((ub >> 16) & 1u);
    return (ua >> 16) | (ub & 0xFFFF0000u);
#endif
}

// MODE 0: pg  = b1 + Target @ W1[woff..]    (rows = graphs, gather via offsets)
// MODE 1: pg += Graph @ W1[woff..]          (rows = graphs)
// MODE 2: out = relu(Node @ W1[0..128] + pg[graph]) . W2 + b2   (rows = nodes)
template <int MODE>
__global__ __launch_bounds__(256, 4)      // cap at 256 VGPR/wave, 4 waves/SIMD
void fused_gemm(const float* __restrict__ Arows,
                const int*   __restrict__ offsets,
                const float* __restrict__ W1,
                int woff,
                const float* __restrict__ initvec,   // MODE0: b1, MODE1/2: pg
                const float* __restrict__ W2,
                const float* __restrict__ b2,
                float* __restrict__ outp,            // MODE0/1: pg, MODE2: logits
                int nStrips, int cnt)
{
    // W1 slice packed into bf16 B-fragment layout:
    // entry (k, t, lane): 16 bf16 = W1[woff + 32k + 8*(lane>=16) + {0..7, 16..23}][16t + (lane&15)]
    __shared__ __align__(32) unsigned ldsb[KSTEPS * NTILE * 32 * 8];   // 64 KB

    const int lane  = threadIdx.x & 31;
    const int laneq = lane & 15;
    const int hi    = lane >> 4;          // 0 or 1

    for (int e = threadIdx.x; e < KSTEPS * NTILE * 32; e += blockDim.x) {
        int el  = e & 31;
        int t   = (e >> 5) & (NTILE - 1);
        int k   = e >> 9;
        int col = t * 16 + (el & 15);
        int kb  = woff + k * 32 + ((el >> 4) << 3);
        const float* wp = W1 + (size_t)kb * HDIM + col;
        unsigned* dst = &ldsb[e * 8];
        #pragma unroll
        for (int j = 0; j < 8; ++j)
            dst[j] = pk_bf16(wp[(2 * j + (j >= 4 ? 8 : 0)) * HDIM], wp[(2 * j + 1 + (j >= 4 ? 8 : 0)) * HDIM]);
    }
    __syncthreads();

    // per-half constant vectors, selected by runtime h via cndmask (no dynamic
    // register-array indexing -> no scratch)
    float w2a[HTILE], w2b[HTILE];
    float b2v = 0.0f;
    if (MODE == 2) {
        #pragma unroll
        for (int t = 0; t < HTILE; ++t) {
            w2a[t] = W2[t * 16 + laneq];
            w2b[t] = W2[(t + HTILE) * 16 + laneq];
        }
        b2v = b2[0];
    }
    float b1a[HTILE], b1b[HTILE];
    if (MODE == 0) {
        #pragma unroll
        for (int t = 0; t < HTILE; ++t) {
            b1a[t] = initvec[t * 16 + laneq];
            b1b[t] = initvec[(t + HTILE) * 16 + laneq];
        }
    }

    const int wid = blockIdx.x * (blockDim.x >> 5) + (threadIdx.x >> 5);
    const int nw  = gridDim.x * (blockDim.x >> 5);

    for (int s = wid; s < nStrips; s += nw) {
        const int m0 = s * 16;

        const float* arow;
        if (MODE == 0) {
            int tgt = offsets[m0 + laneq + 1] - 1;     // last node of graph
            arow = Arows + (size_t)tgt * DIM;
        } else {
            arow = Arows + (size_t)(m0 + laneq) * DIM;
        }

        // A fragments: lane holds row (m0 + lane&15); 16-bit A layout = two
        // contiguous 8-elem chunks at K = 32k + 8*hi and K = 32k + 16 + 8*hi.
        // Loaded once per strip, reused by both column halves.
        v16bf afrag[KSTEPS];
        const float* ab = arow + hi * 8;
        #pragma unroll
        for (int k = 0; k < KSTEPS; ++k) {
            const float4 f0 = *(const float4*)(ab + k * 32);
            const float4 f1 = *(const float4*)(ab + k * 32 + 4);
            const float4 f2 = *(const float4*)(ab + k * 32 + 16);
            const float4 f3 = *(const float4*)(ab + k * 32 + 20);
            v8u au;
            au[0] = pk_bf16(f0.x, f0.y); au[1] = pk_bf16(f0.z, f0.w);
            au[2] = pk_bf16(f1.x, f1.y); au[3] = pk_bf16(f1.z, f1.w);
            au[4] = pk_bf16(f2.x, f2.y); au[5] = pk_bf16(f2.z, f2.w);
            au[6] = pk_bf16(f3.x, f3.y); au[7] = pk_bf16(f3.z, f3.w);
            afrag[k] = __builtin_bit_cast(v16bf, au);
        }

        const float* pgrow = (MODE == 2)
            ? initvec + (size_t)(m0 / cnt) * HDIM      // strip lies within one graph
            : initvec;

        float partial[8];
        if (MODE == 2) {
            #pragma unroll
            for (int r = 0; r < 8; ++r) partial[r] = 0.0f;
        }

        // two sequential halves of H: keep only 8 acc tiles (64 VGPRs) live
        #pragma unroll 1
        for (int h = 0; h < 2; ++h) {
            v8f acc[HTILE];
            if (MODE == 0) {
                #pragma unroll
                for (int t = 0; t < HTILE; ++t) {
                    float v = h ? b1b[t] : b1a[t];
                    acc[t] = (v8f){v, v, v, v, v, v, v, v};
                }
            } else if (MODE == 1) {
                #pragma unroll
                for (int t = 0; t < HTILE; ++t)
                    #pragma unroll
                    for (int r = 0; r < 8; ++r)
                        acc[t][r] = initvec[(size_t)(m0 + hi * 8 + r) * HDIM +
                                            (h * HTILE + t) * 16 + laneq];
            } else {
                #pragma unroll
                for (int t = 0; t < HTILE; ++t) {
                    float v = pgrow[(h * HTILE + t) * 16 + laneq];
                    acc[t] = (v8f){v, v, v, v, v, v, v, v};
                }
            }

            #pragma unroll
            for (int k = 0; k < KSTEPS; ++k) {
                #pragma unroll
                for (int t = 0; t < HTILE; ++t) {
                    const v8u bu = *(const v8u*)
                        &ldsb[((k * NTILE + h * HTILE + t) * 32 + lane) * 8];
                    v16bf bfrag = __builtin_bit_cast(v16bf, bu);
                    acc[t] = __builtin_amdgcn_wmma_f32_16x16x32_bf16(
                        false, afrag[k], false, bfrag, (short)0, acc[t], false, false);
                }
            }

            if (MODE == 2) {
                // relu + dot(W2) partial; additive across halves
                #pragma unroll
                for (int r = 0; r < 8; ++r)
                    #pragma unroll
                    for (int t = 0; t < HTILE; ++t) {
                        float wv = h ? w2b[t] : w2a[t];
                        partial[r] += fmaxf(acc[t][r], 0.0f) * wv;
                    }
            } else {
                #pragma unroll
                for (int t = 0; t < HTILE; ++t)
                    #pragma unroll
                    for (int r = 0; r < 8; ++r)
                        outp[(size_t)(m0 + hi * 8 + r) * HDIM +
                             (h * HTILE + t) * 16 + laneq] = acc[t][r];
            }
        }

        if (MODE == 2) {
            // reduce the 16 lanes of each half (halves hold disjoint node rows)
            #pragma unroll
            for (int r = 0; r < 8; ++r) {
                float p = partial[r];
                p += __shfl_xor(p, 1);
                p += __shfl_xor(p, 2);
                p += __shfl_xor(p, 4);
                p += __shfl_xor(p, 8);
                if (laneq == 0) outp[m0 + hi * 8 + r] = p + b2v;
            }
        }
    }
}

extern "C" void kernel_launch(void* const* d_in, const int* in_sizes, int n_in,
                              void* d_out, int out_size, void* d_ws, size_t ws_size,
                              hipStream_t stream) {
    const float* node_emb  = (const float*)d_in[0];
    const float* graph_emb = (const float*)d_in[1];
    const int*   offsets   = (const int*)  d_in[3];
    const float* W1        = (const float*)d_in[4];
    const float* b1        = (const float*)d_in[5];
    const float* W2        = (const float*)d_in[6];
    const float* b2        = (const float*)d_in[7];
    float*       out       = (float*)d_out;
    float*       pg        = (float*)d_ws;          // [B, 256] per-graph bias term

    const int Ntot = in_sizes[0] / DIM;             // 524288 nodes
    const int Bg   = in_sizes[1] / DIM;             // 8192 graphs
    const int cnt  = Ntot / Bg;                     // 64 nodes/graph (uniform)

    const int sA = Bg   / 16;                       // 512 graph strips
    const int sB = Ntot / 16;                       // 32768 node strips
    dim3 blk(256);                                  // 8 wave32 per block

    fused_gemm<0><<<(sA + 7) / 8, blk, 0, stream>>>(node_emb,  offsets, W1, DIM,
                                                    b1, nullptr, nullptr, pg, sA, cnt);
    fused_gemm<1><<<(sA + 7) / 8, blk, 0, stream>>>(graph_emb, nullptr, W1, 2 * DIM,
                                                    pg, nullptr, nullptr, pg, sA, cnt);
    fused_gemm<2><<<512, blk, 0, stream>>>(node_emb, nullptr, W1, 0,
                                           pg, W2, b2, out, sB, cnt);
}
